// Attention_29205777613771
// MI455X (gfx1250) — matrix operand order repaired and checked by reading, not compiled, not run on hardware
//
#include <hip/hip_runtime.h>

// MI455X / gfx1250: wave32, WMMA bf16 16x16x32, f32 accumulate,
// async global->LDS staging (ASYNCcnt) for K/V tiles.
typedef __attribute__((ext_vector_type(16))) __bf16 v16bf;
typedef __attribute__((ext_vector_type(8)))  float  v8f;
typedef __attribute__((ext_vector_type(4)))  int    v4i;

typedef __attribute__((address_space(1))) v4i as1_v4i;   // global (__device__)
typedef __attribute__((address_space(3))) v4i as3_v4i;   // LDS (__shared__)

union ATile {
  v16bf v;
  unsigned short u[16];
  uint4 q[2];
};

__device__ __forceinline__ unsigned short f2bf(float f) {
  unsigned int u = __float_as_uint(f);
  u += 0x7FFFu + ((u >> 16) & 1u);   // round-to-nearest-even
  return (unsigned short)(u >> 16);
}

__device__ __forceinline__ void pack8(const float* __restrict__ p, unsigned short* u) {
  float4 a = *(const float4*)p;
  float4 b = *(const float4*)(p + 4);
  u[0] = f2bf(a.x); u[1] = f2bf(a.y); u[2] = f2bf(a.z); u[3] = f2bf(a.w);
  u[4] = f2bf(b.x); u[5] = f2bf(b.y); u[6] = f2bf(b.z); u[7] = f2bf(b.w);
}

#define WMMA_BF16(A, B, Cacc) \
  __builtin_amdgcn_wmma_f32_16x16x32_bf16(false, (A), false, (B), (short)0, (Cacc), false, false)

// --- gfx1250 async global->LDS copy (ASYNCcnt) with sync fallback ----------
#if __has_builtin(__builtin_amdgcn_global_load_async_to_lds_b128)
#define HAVE_ASYNC 1
__device__ __forceinline__ void async_copy_b128(const unsigned short* g, unsigned short* l) {
  __builtin_amdgcn_global_load_async_to_lds_b128(
      (as1_v4i*)(void*)const_cast<unsigned short*>(g),
      (as3_v4i*)(void*)l, 0, 0);
}
__device__ __forceinline__ void wait_async() {
#if __has_builtin(__builtin_amdgcn_s_wait_asynccnt)
  __builtin_amdgcn_s_wait_asynccnt(0);
#else
  asm volatile("s_wait_asynccnt 0x0" ::: "memory");
#endif
}
#else
#define HAVE_ASYNC 0
__device__ __forceinline__ void async_copy_b128(const unsigned short* g, unsigned short* l) {
  *(uint4*)l = *(const uint4*)g;   // synchronous fallback
}
__device__ __forceinline__ void wait_async() {}
#endif

// Sizes (fixed by the reference)
#define BH   64      // B*H
#define SEQ  2048
#define CH   64
#define C3   192

// ---------------------------------------------------------------------------
// Kernel 0: one-shot W f32 -> bf16.
// ---------------------------------------------------------------------------
__global__ void __launch_bounds__(256)
w_to_bf16_kernel(const float* __restrict__ W, unsigned short* __restrict__ Wb) {
  const int i = blockIdx.x * 256 + threadIdx.x;   // 12288 elements
  if (i < C3 * CH) Wb[i] = f2bf(W[i]);
}

// ---------------------------------------------------------------------------
// Kernel 1: fused QKV projection (x[BH*SEQ,64] @ W^T[64,192]) in bf16 WMMA.
// One wave per 16 sequence rows. Q,K row-major bf16; V transposed [C][N].
// ---------------------------------------------------------------------------
__global__ void __launch_bounds__(32)
qkv_wmma_kernel(const float* __restrict__ x, const unsigned short* __restrict__ Wb,
                unsigned short* __restrict__ Qb, unsigned short* __restrict__ Kb,
                unsigned short* __restrict__ Vt) {
  const int lane = threadIdx.x & 31;
  const int g = lane >> 4, ln = lane & 15;
  const int tile = blockIdx.x;          // BH * (SEQ/16) tiles
  const int bh = tile >> 7;             // SEQ/16 = 128
  const int n0 = (tile & 127) << 4;

  // A operand: 16 rows x 64 channels as two K=32 chunks (ISA 16-bit A layout:
  // lane = row, halves 0..7 -> k = g*8+i, halves 8..15 -> k = 16+g*8+(i-8))
  ATile a[2];
  const float* xrow = x + ((size_t)bh * SEQ + n0 + ln) * CH;
#pragma unroll
  for (int kc = 0; kc < 2; ++kc) {
    pack8(xrow + kc * 32 + g * 8, a[kc].u);
    pack8(xrow + kc * 32 + 16 + g * 8, a[kc].u + 8);
  }

#pragma unroll
  for (int t = 0; t < 12; ++t) {        // 12 tiles of 16 output channels
    const int outch = t * 16 + ln;      // B layout: lane = column, k = g*8..+7 and 16+g*8..+7 per group
    const unsigned short* wrow = Wb + (size_t)outch * CH;
    v8f acc = {};
#pragma unroll
    for (int kc = 0; kc < 2; ++kc) {
      ATile b;
      b.q[0] = *(const uint4*)(wrow + kc * 32 + g * 8);
      b.q[1] = *(const uint4*)(wrow + kc * 32 + 16 + g * 8);
      acc = WMMA_BF16(a[kc].v, b.v, acc);
    }
    // C/D layout: lane holds (m = r + 8*g, n = ln), r = 0..7
    if (t < 4) {
      const int c = t * 16 + ln;
#pragma unroll
      for (int r = 0; r < 8; ++r)
        Qb[((size_t)bh * SEQ + n0 + r + 8 * g) * CH + c] = f2bf(acc[r]);
    } else if (t < 8) {
      const int c = (t - 4) * 16 + ln;
#pragma unroll
      for (int r = 0; r < 8; ++r)
        Kb[((size_t)bh * SEQ + n0 + r + 8 * g) * CH + c] = f2bf(acc[r]);
    } else {
      const int c = (t - 8) * 16 + ln;  // V stored transposed: [bh][c][n]
#pragma unroll
      for (int r = 0; r < 8; ++r)
        Vt[((size_t)bh * CH + c) * SEQ + n0 + r + 8 * g] = f2bf(acc[r]);
    }
  }
}

// ---------------------------------------------------------------------------
// Kernel 2: flash attention. 4 waves/block, each wave owns 32 query rows
// (block = 128 rows). K/V 32-key chunks are staged global->LDS once per
// block with async-to-LDS double buffering and shared by all 4 waves.
// 16 WMMAs per wave per chunk; online softmax in f32; head-mix on store.
// ---------------------------------------------------------------------------
__global__ void __launch_bounds__(128)
flash_attn_wmma_kernel(const unsigned short* __restrict__ Qb,
                       const unsigned short* __restrict__ Kb,
                       const unsigned short* __restrict__ Vt,
                       float* __restrict__ out) {
  __shared__ __align__(16) unsigned short k_lds[2][32 * CH];   // [buf][key][ch]
  __shared__ __align__(16) unsigned short v_lds[2][CH * 32];   // [buf][ch][key]
  __shared__ __align__(16) unsigned short p_lds[4][2][16 * 32];// per-wave P bounce

  const int tid = threadIdx.x;
  const int lane = tid & 31;
  const int w = tid >> 5;                 // wave 0..3
  const int g = lane >> 4, ln = lane & 15;
  const int blk = blockIdx.x;
  const int bh = blk >> 4;                // 16 blocks of 128 rows per head
  const int q0 = ((blk & 15) << 7) + w * 32;
  const int bb = bh >> 3, hh = bh & 7;

  const unsigned short* kbase = Kb + (size_t)bh * SEQ * CH;
  const unsigned short* vbase = Vt + (size_t)bh * CH * SEQ;

  // Q as A operand (2 q-tiles x 2 K-chunks), resident for the whole block
  ATile qa[2][2];
#pragma unroll
  for (int qt = 0; qt < 2; ++qt) {
    const unsigned short* qrow = Qb + ((size_t)bh * SEQ + q0 + qt * 16 + ln) * CH;
#pragma unroll
    for (int kc = 0; kc < 2; ++kc) {
      qa[qt][kc].q[0] = *(const uint4*)(qrow + kc * 32 + g * 8);
      qa[qt][kc].q[1] = *(const uint4*)(qrow + kc * 32 + 16 + g * 8);
    }
  }

  float mi[2][8], li[2][8];
  v8f acc[2][4];
  {
    v8f z = {};
#pragma unroll
    for (int qt = 0; qt < 2; ++qt)
#pragma unroll
      for (int ct = 0; ct < 4; ++ct) acc[qt][ct] = z;
  }
#pragma unroll
  for (int qt = 0; qt < 2; ++qt)
#pragma unroll
    for (int r = 0; r < 8; ++r) { mi[qt][r] = -3.0e38f; li[qt][r] = 0.0f; }

  // issue async copy of chunk j into buffer buf: 128 threads x 2 b128 each side
  auto issue_copy = [&](int buf, int j) {
#pragma unroll
    for (int it = 0; it < 2; ++it) {
      const int e = tid + it * 128;           // 0..255 b128 units
      const int krow = e >> 3, kc8 = (e & 7) * 8;
      const int vrow = e >> 2, vk8 = (e & 3) * 8;
      async_copy_b128(kbase + (size_t)(j + krow) * CH + kc8,
                      &k_lds[buf][krow * CH + kc8]);
      async_copy_b128(vbase + (size_t)vrow * SEQ + j + vk8,
                      &v_lds[buf][vrow * 32 + vk8]);
    }
  };

  issue_copy(0, 0);                           // pipeline prologue

  for (int j = 0; j < SEQ; j += 32) {
    const int buf = (j >> 5) & 1;
    wait_async();
    __syncthreads();                          // chunk `buf` visible to all waves
    if (j + 32 < SEQ) issue_copy(buf ^ 1, j + 32);

    // ---- per q-tile: S = Q K^T from LDS K tiles, online softmax, P -> LDS ----
#pragma unroll
    for (int qt = 0; qt < 2; ++qt) {
      v8f s0 = {}, s1 = {};
#pragma unroll
      for (int kc = 0; kc < 2; ++kc) {
        ATile kb0, kb1;   // lane = key column, channels g*8..+7 and 16+g*8..+7
        const unsigned short* kr0 = &k_lds[buf][(ln)      * CH + kc * 32 + g * 8];
        const unsigned short* kr1 = &k_lds[buf][(16 + ln) * CH + kc * 32 + g * 8];
        kb0.q[0] = *(const uint4*)kr0; kb0.q[1] = *(const uint4*)(kr0 + 16);
        kb1.q[0] = *(const uint4*)kr1; kb1.q[1] = *(const uint4*)(kr1 + 16);
        s0 = WMMA_BF16(qa[qt][kc].v, kb0.v, s0);
        s1 = WMMA_BF16(qa[qt][kc].v, kb1.v, s1);
      }
      float alpha[8];
#pragma unroll
      for (int r = 0; r < 8; ++r) {
        float v0 = s0[r] * 0.125f, v1 = s1[r] * 0.125f;   // scale = C^-0.5
        float mx = fmaxf(v0, v1);
        mx = fmaxf(mx, __shfl_xor(mx, 1, 32));
        mx = fmaxf(mx, __shfl_xor(mx, 2, 32));
        mx = fmaxf(mx, __shfl_xor(mx, 4, 32));
        mx = fmaxf(mx, __shfl_xor(mx, 8, 32));
        float mn = fmaxf(mi[qt][r], mx);
        alpha[r] = __expf(mi[qt][r] - mn);
        mi[qt][r] = mn;
        float p0 = __expf(v0 - mn);
        float p1 = __expf(v1 - mn);
        s0[r] = p0; s1[r] = p1;
        float rs = p0 + p1;
        rs += __shfl_xor(rs, 1, 32);
        rs += __shfl_xor(rs, 2, 32);
        rs += __shfl_xor(rs, 4, 32);
        rs += __shfl_xor(rs, 8, 32);
        li[qt][r] = li[qt][r] * alpha[r] + rs;
      }
#pragma unroll
      for (int ct = 0; ct < 4; ++ct)
#pragma unroll
        for (int r = 0; r < 8; ++r) acc[qt][ct][r] *= alpha[r];

      // P in C/D layout -> per-wave LDS (m, k) tile (wave-private, DS in-order)
#pragma unroll
      for (int r = 0; r < 8; ++r) {
        const int m = r + 8 * g;
        p_lds[w][qt][m * 32 + ln] = f2bf(s0[r]);
        p_lds[w][qt][m * 32 + 16 + ln] = f2bf(s1[r]);
      }
    }

    // ---- read P back in A layout ----
    ATile pa[2];
#pragma unroll
    for (int qt = 0; qt < 2; ++qt) {
      pa[qt].q[0] = *(const uint4*)(&p_lds[w][qt][0] + ln * 32 + g * 8);
      pa[qt].q[1] = *(const uint4*)(&p_lds[w][qt][0] + ln * 32 + 16 + g * 8);
    }

    // ---- O += P V : V tiles from LDS, shared across both q-tiles ----
#pragma unroll
    for (int ct = 0; ct < 4; ++ct) {
      ATile vb;   // lane = channel column, keys g*8..+7 and 16+g*8..+7
      const unsigned short* vr = &v_lds[buf][(ct * 16 + ln) * 32 + g * 8];
      vb.q[0] = *(const uint4*)vr;
      vb.q[1] = *(const uint4*)(vr + 16);
      acc[0][ct] = WMMA_BF16(pa[0].v, vb.v, acc[0][ct]);
      acc[1][ct] = WMMA_BF16(pa[1].v, vb.v, acc[1][ct]);
    }
    __syncthreads();   // all waves done with `buf` before it is refilled
  }

  // ---- epilogue: divide by row sums, fold in head-mixing permutation ----
  // final[b][c/8][n][h*8 + c%8] = attn_out[b][h][n][c]
#pragma unroll
  for (int qt = 0; qt < 2; ++qt)
#pragma unroll
    for (int r = 0; r < 8; ++r) {
      const float inv = 1.0f / li[qt][r];
      const int m = q0 + qt * 16 + r + 8 * g;
#pragma unroll
      for (int ct = 0; ct < 4; ++ct) {
        const int c = ct * 16 + ln;
        const int j2 = c >> 3, cr = c & 7;
        out[(((size_t)bb * 8 + j2) * SEQ + m) * CH + hh * 8 + cr] = acc[qt][ct][r] * inv;
      }
    }
}

// ---------------------------------------------------------------------------
extern "C" void kernel_launch(void* const* d_in, const int* in_sizes, int n_in,
                              void* d_out, int out_size, void* d_ws, size_t ws_size,
                              hipStream_t stream) {
  (void)in_sizes; (void)n_in; (void)out_size; (void)ws_size;
  const float* x = (const float*)d_in[0];      // [8,8,2048,64] f32
  const float* W = (const float*)d_in[1];      // [192,64] f32
  float* out = (float*)d_out;                  // [8,8,2048,64] f32

  const size_t perbuf = (size_t)BH * SEQ * CH; // bf16 elements
  unsigned short* Qb = (unsigned short*)d_ws;
  unsigned short* Kb = Qb + perbuf;
  unsigned short* Vt = Kb + perbuf;            // V stored [bh][c][n]
  unsigned short* Wb = Vt + perbuf;            // 24 KB bf16 copy of W

  w_to_bf16_kernel<<<(C3 * CH + 255) / 256, 256, 0, stream>>>(W, Wb);
  qkv_wmma_kernel<<<BH * (SEQ / 16), 32, 0, stream>>>(x, Wb, Qb, Kb, Vt);
  flash_attn_wmma_kernel<<<BH * (SEQ / 128), 128, 0, stream>>>(Qb, Kb, Vt, out);
}
